// MambaFractralBlock_40261023433410
// MI455X (gfx1250) — matrix operand order repaired and checked by reading, not compiled, not run on hardware
//
#include <hip/hip_runtime.h>
#include <hip/hip_bf16.h>

typedef __attribute__((ext_vector_type(16))) _Float16 v16h;
typedef __attribute__((ext_vector_type(8)))  _Float16 v8h;
typedef __attribute__((ext_vector_type(8)))  float    v8f;

#define Lq   1024
#define Dq   1024
#define EDq  2048
#define XZW  4096   // 2*ED
#define RNq  96     // R + 2N
#define Rq   64
#define Nst  16

__device__ __forceinline__ float silu_f(float x) { return x / (1.0f + __expf(-x)); }
__device__ __forceinline__ float softplus_f(float x) {
    return (x > 20.0f) ? x : __logf(1.0f + __expf(x));
}

// ---------------------------------------------------------------------------
// Convert f32 weight matrix B (K x N, row-major) into f16 pre-swizzled WMMA
// B-fragment layout:  P[((kt*(N/16) + nt)*32 + lane)*16 + i]
//   lane <  16 : element i = B[kt*32 + (i&7) + (i>=8 ? 16 : 0)][nt*16 + lane]
//   lane >= 16 : element i = B[kt*32 + 8 + (i&7) + (i>=8 ? 16 : 0)][nt*16 + lane-16]
// so each lane's 16 halves are contiguous (2 x b128 loads in the GEMM).
// ---------------------------------------------------------------------------
__global__ void pack_b_f16_kernel(const float* __restrict__ B,
                                  _Float16* __restrict__ P, int K, int N) {
    int idx = blockIdx.x * blockDim.x + threadIdx.x;
    if (idx >= K * N) return;
    int i    = idx & 15;
    int lane = (idx >> 4) & 31;
    int tile = idx >> 9;                 // kt*(N/16) + nt
    int ntn  = N >> 4;
    int kt   = tile / ntn;
    int nt   = tile - kt * ntn;
    int n = nt * 16 + (lane & 15);
    int k = kt * 32 + ((lane >= 16) ? 8 : 0) + (i & 7) + ((i >= 8) ? 16 : 0);
    P[idx] = (_Float16)B[(size_t)k * N + n];
}

__global__ void copy_f32_kernel(const float* __restrict__ src,
                                float* __restrict__ dst, int n) {
    int i = blockIdx.x * blockDim.x + threadIdx.x;
    if (i < n) dst[i] = src[i];
}

// ---------------------------------------------------------------------------
// RMSNorm row -> f16 output (GEMM A operand)
// ---------------------------------------------------------------------------
__global__ __launch_bounds__(256)
void rmsnorm_f16_kernel(const float* __restrict__ x, const float* __restrict__ w,
                        _Float16* __restrict__ out, int d) {
    __shared__ float red[256];
    int row = blockIdx.x;
    const float* xr = x + (size_t)row * d;
    float s = 0.0f;
    for (int i = threadIdx.x; i < d; i += 256) { float v = xr[i]; s += v * v; }
    red[threadIdx.x] = s; __syncthreads();
    for (int o = 128; o > 0; o >>= 1) {
        if (threadIdx.x < o) red[threadIdx.x] += red[threadIdx.x + o];
        __syncthreads();
    }
    float inv = rsqrtf(red[0] / (float)d + 1e-5f);
    for (int i = threadIdx.x; i < d; i += 256)
        out[(size_t)row * d + i] = (_Float16)(xr[i] * inv * w[i]);
}

// ---------------------------------------------------------------------------
// WMMA GEMM: C(f32, MxN) = A(f16, MxK, lda) * Bpack(pre-swizzled f16)
// 8 waves/block; each wave computes a 16x32 output (2 N-tiles, one A frag).
// mode 0: store acc
// mode 1: store softplus(acc + bias[col])
// mode 2: store acc + 2*resid[idx]           (fused Mamba double-residual)
// ---------------------------------------------------------------------------
__global__ __launch_bounds__(256)
void gemm_wmma_f16_kernel(const _Float16* __restrict__ A, int lda,
                          const _Float16* __restrict__ Bpack,
                          float* __restrict__ C, int ldc,
                          int M, int N, int K,
                          int mode, const float* __restrict__ bias,
                          const float* __restrict__ resid) {
    const int lane = threadIdx.x & 31;
    const int wave = threadIdx.x >> 5;
    const int n0 = blockIdx.x * 32;                  // two 16-wide N tiles
    const int m0 = (blockIdx.y * 8 + wave) * 16;
    if (m0 >= M || n0 >= N) return;                  // wave-uniform

    const int  mr = lane & 15;
    const bool hi = lane >= 16;
    const int  ntn = N >> 4;
    const int  nt0 = n0 >> 4;

    // A fragment rows: lane -> row m0+mr, K-phase 0/8 (ISA 16-bit A layout)
    const _Float16* Arow = A + (size_t)(m0 + mr) * lda + (hi ? 8 : 0);
    // Packed B: per-lane contiguous 16 halves per (ktile, ntile)
    const _Float16* Bp0 = Bpack + ((size_t)nt0 * 32 + lane) * 16;
    const size_t kstride = (size_t)ntn * 512;        // halves per k-tile row

    v8f acc0 = {}, acc1 = {};
    for (int k0 = 0; k0 < K; k0 += 32) {
        __builtin_prefetch(Arow + k0 + 128, 0, 1);

        v8h alo = *(const v8h*)(Arow + k0);          // K = kb+0 .. kb+7
        v8h ahi = *(const v8h*)(Arow + k0 + 16);     // K = kb+16 .. kb+23
        v16h a;
#pragma unroll
        for (int i = 0; i < 8; ++i) { a[i] = alo[i]; a[8 + i] = ahi[i]; }

        const _Float16* Bk = Bp0 + (size_t)(k0 >> 5) * kstride;
        v8h b0l = *(const v8h*)(Bk);
        v8h b0h = *(const v8h*)(Bk + 8);
        v8h b1l = *(const v8h*)(Bk + 512);
        v8h b1h = *(const v8h*)(Bk + 512 + 8);
        v16h b0, b1;
#pragma unroll
        for (int i = 0; i < 8; ++i) {
            b0[i] = b0l[i]; b0[8 + i] = b0h[i];
            b1[i] = b1l[i]; b1[8 + i] = b1h[i];
        }

        acc0 = __builtin_amdgcn_wmma_f32_16x16x32_f16(
            false, a, false, b0, (short)0, acc0, false, false);
        acc1 = __builtin_amdgcn_wmma_f32_16x16x32_f16(
            false, a, false, b1, (short)0, acc1, false, false);
    }

#pragma unroll
    for (int t = 0; t < 2; ++t) {
        const v8f& acc = t ? acc1 : acc0;
        int nc = n0 + t * 16 + mr;
        float bcol = (mode == 1) ? bias[nc] : 0.0f;
#pragma unroll
        for (int r = 0; r < 8; ++r) {
            int row = m0 + r + (hi ? 8 : 0);
            size_t idx = (size_t)row * ldc + nc;
            float v = acc[r];
            if (mode == 1)      v = softplus_f(v + bcol);
            else if (mode == 2) v = v + 2.0f * resid[idx];
            C[idx] = v;
        }
    }
}

// ---------------------------------------------------------------------------
// Depthwise causal conv (K=4) + bias + SiLU.  xs = xz[:, :ED].
// ---------------------------------------------------------------------------
__global__ __launch_bounds__(256)
void conv_silu_kernel(const float* __restrict__ xz,
                      const float* __restrict__ cw, const float* __restrict__ cb,
                      float* __restrict__ xc, _Float16* __restrict__ xch) {
    int idx = blockIdx.x * blockDim.x + threadIdx.x;   // over L*ED
    if (idx >= Lq * EDq) return;
    int e = idx % EDq, l = idx / EDq;
    float acc = cb[e];
#pragma unroll
    for (int k = 0; k < 4; ++k) {
        int ls = l - 3 + k;
        if (ls >= 0) acc += cw[e * 4 + k] * xz[(size_t)ls * XZW + e];
    }
    float v = silu_f(acc);
    xc[idx] = v;
    xch[idx] = (_Float16)v;
}

// ---------------------------------------------------------------------------
// Extract dt columns [0,64) of dbc into f16 (GEMM A operand)
// ---------------------------------------------------------------------------
__global__ void extract_dt_f16_kernel(const float* __restrict__ dbc,
                                      _Float16* __restrict__ dth) {
    int idx = blockIdx.x * blockDim.x + threadIdx.x;   // L*R
    if (idx >= Lq * Rq) return;
    int c = idx % Rq, l = idx / Rq;
    dth[idx] = (_Float16)dbc[(size_t)l * RNq + c];
}

// ---------------------------------------------------------------------------
// Fused selective scan. One thread per channel e, 16 states in registers.
// ---------------------------------------------------------------------------
__global__ __launch_bounds__(256)
void scan_kernel(const float* __restrict__ delta, const float* __restrict__ dbc,
                 const float* __restrict__ xc, const float* __restrict__ xz,
                 const float* __restrict__ A_log, const float* __restrict__ Dp,
                 _Float16* __restrict__ yh) {
    int e = blockIdx.x * blockDim.x + threadIdx.x;
    if (e >= EDq) return;
    float An[Nst], h[Nst];
#pragma unroll
    for (int n = 0; n < Nst; ++n) {
        An[n] = -__expf(A_log[(size_t)e * Nst + n]);
        h[n] = 0.0f;
    }
    const float dpe = Dp[e];
    for (int l = 0; l < Lq; ++l) {
        float dl = delta[(size_t)l * EDq + e];
        float xv = xc[(size_t)l * EDq + e];
        const float* row = dbc + (size_t)l * RNq;   // [dt(64) | B(16) | C(16)]
        float dx = dl * xv;
        float y = 0.0f;
#pragma unroll
        for (int n = 0; n < Nst; ++n) {
            float dA = __expf(dl * An[n]);
            h[n] = dA * h[n] + dx * row[Rq + n];
            y += h[n] * row[Rq + Nst + n];
        }
        y += dpe * xv;
        float z = xz[(size_t)l * XZW + EDq + e];
        yh[(size_t)l * EDq + e] = (_Float16)(y * silu_f(z));
    }
}

// ---------------------------------------------------------------------------
// Final LayerNorm
// ---------------------------------------------------------------------------
__global__ __launch_bounds__(256)
void layernorm_kernel(const float* __restrict__ x, const float* __restrict__ w,
                      const float* __restrict__ b, float* __restrict__ out, int d) {
    __shared__ float red[256];
    int row = blockIdx.x;
    const float* xr = x + (size_t)row * d;
    float s = 0.0f;
    for (int i = threadIdx.x; i < d; i += 256) s += xr[i];
    red[threadIdx.x] = s; __syncthreads();
    for (int o = 128; o > 0; o >>= 1) {
        if (threadIdx.x < o) red[threadIdx.x] += red[threadIdx.x + o];
        __syncthreads();
    }
    float mu = red[0] / (float)d;
    __syncthreads();
    float v = 0.0f;
    for (int i = threadIdx.x; i < d; i += 256) { float t = xr[i] - mu; v += t * t; }
    red[threadIdx.x] = v; __syncthreads();
    for (int o = 128; o > 0; o >>= 1) {
        if (threadIdx.x < o) red[threadIdx.x] += red[threadIdx.x + o];
        __syncthreads();
    }
    float inv = rsqrtf(red[0] / (float)d + 1e-5f);
    for (int i = threadIdx.x; i < d; i += 256)
        out[(size_t)row * d + i] = (xr[i] - mu) * inv * w[i] + b[i];
}

// ---------------------------------------------------------------------------
// Host launcher
// ---------------------------------------------------------------------------
extern "C" void kernel_launch(void* const* d_in, const int* in_sizes, int n_in,
                              void* d_out, int out_size, void* d_ws, size_t ws_size,
                              hipStream_t stream) {
    const float* x_in  = (const float*)d_in[0];
    const float* rmsw  = (const float*)d_in[1];
    const float* Win   = (const float*)d_in[2];
    const float* convw = (const float*)d_in[3];
    const float* convb = (const float*)d_in[4];
    const float* Wx    = (const float*)d_in[5];
    const float* Wdt   = (const float*)d_in[6];
    const float* bdt   = (const float*)d_in[7];
    const float* Alog  = (const float*)d_in[8];
    const float* Dp    = (const float*)d_in[9];
    const float* Wout  = (const float*)d_in[10];
    const float* lnw   = (const float*)d_in[11];
    const float* lnb   = (const float*)d_in[12];

    // workspace carve (256B aligned)
    char* p = (char*)d_ws;
    auto carve = [&](size_t bytes) -> void* {
        void* r = (void*)p;
        p += (bytes + 255) & ~(size_t)255;
        return r;
    };
    _Float16* Win_p  = (_Float16*)carve((size_t)Dq * XZW * 2);
    _Float16* Wx_p   = (_Float16*)carve((size_t)EDq * RNq * 2);
    _Float16* Wdt_p  = (_Float16*)carve((size_t)Rq * EDq * 2);
    _Float16* Wout_p = (_Float16*)carve((size_t)EDq * Dq * 2);
    _Float16* un_h   = (_Float16*)carve((size_t)Lq * Dq * 2);
    float*    xz     = (float*)   carve((size_t)Lq * XZW * 4);
    float*    xc     = (float*)   carve((size_t)Lq * EDq * 4);
    _Float16* xc_h   = (_Float16*)carve((size_t)Lq * EDq * 2);
    float*    dbc    = (float*)   carve((size_t)Lq * RNq * 4);
    _Float16* dt_h   = (_Float16*)carve((size_t)Lq * Rq * 2);
    float*    delta  = (float*)   carve((size_t)Lq * EDq * 4);
    _Float16* y_h    = (_Float16*)carve((size_t)Lq * EDq * 2);
    float*    xbuf   = (float*)   carve((size_t)Lq * Dq * 4);

    const int T = 256;
    // one-time weight convert + pre-swizzle into WMMA B-fragment layout
    pack_b_f16_kernel<<<(Dq * XZW + T - 1) / T, T, 0, stream>>>(Win,  Win_p,  Dq,  XZW);
    pack_b_f16_kernel<<<(EDq * RNq + T - 1) / T, T, 0, stream>>>(Wx,   Wx_p,   EDq, RNq);
    pack_b_f16_kernel<<<(Rq * EDq + T - 1) / T, T, 0, stream>>>(Wdt,  Wdt_p,  Rq,  EDq);
    pack_b_f16_kernel<<<(EDq * Dq + T - 1) / T, T, 0, stream>>>(Wout, Wout_p, EDq, Dq);
    copy_f32_kernel<<<(Lq * Dq + T - 1) / T, T, 0, stream>>>(x_in, xbuf, Lq * Dq);

    for (int layer = 0; layer < 6; ++layer) {
        // 1. RMSNorm -> f16
        rmsnorm_f16_kernel<<<Lq, T, 0, stream>>>(xbuf, rmsw, un_h, Dq);
        // 2. xz = un @ W_in   (1024x1024 x 1024x4096)
        gemm_wmma_f16_kernel<<<dim3(XZW / 32, Lq / 128), T, 0, stream>>>(
            un_h, Dq, Win_p, xz, XZW, Lq, XZW, Dq, 0, nullptr, nullptr);
        // 3. depthwise conv + SiLU
        conv_silu_kernel<<<(Lq * EDq + T - 1) / T, T, 0, stream>>>(
            xz, convw, convb, xc, xc_h);
        // 4. dbc = xc @ W_x   (1024x2048 x 2048x96)
        gemm_wmma_f16_kernel<<<dim3(RNq / 32, Lq / 128), T, 0, stream>>>(
            xc_h, EDq, Wx_p, dbc, RNq, Lq, RNq, EDq, 0, nullptr, nullptr);
        // 5. dt -> f16
        extract_dt_f16_kernel<<<(Lq * Rq + T - 1) / T, T, 0, stream>>>(dbc, dt_h);
        // 6. delta = softplus(dt @ W_dt + b_dt)   (1024x64 x 64x2048)
        gemm_wmma_f16_kernel<<<dim3(EDq / 32, Lq / 128), T, 0, stream>>>(
            dt_h, Rq, Wdt_p, delta, EDq, Lq, EDq, Rq, 1, bdt, nullptr);
        // 7. fused selective scan + gate -> y (f16)
        scan_kernel<<<EDq / T, T, 0, stream>>>(delta, dbc, xc, xz, Alog, Dp, y_h);
        // 8. xbuf = y @ W_out + 2*xbuf   (1024x2048 x 2048x1024)
        gemm_wmma_f16_kernel<<<dim3(Dq / 32, Lq / 128), T, 0, stream>>>(
            y_h, EDq, Wout_p, xbuf, Dq, Lq, Dq, EDq, 2, nullptr, xbuf);
    }

    layernorm_kernel<<<Lq, T, 0, stream>>>(xbuf, lnw, lnb, (float*)d_out, Dq);
}
